// ShiftedMambaBlocks_84413287235935
// MI455X (gfx1250) — matrix operand order repaired and checked by reading, not compiled, not run on hardware
//
#include <hip/hip_runtime.h>
#include <cstdint>
#include <cstddef>

// ---------------------------------------------------------------------------
// ShiftedMambaBlocks for MI455X (gfx1250, wave32, WMMA)
// GEMMs: v_wmma_f32_16x16x32_bf16, 2x2 tiles per wave for 2x load amortization
// scan: f32 VALU with v_exp_f32, states in VGPRs, B/C broadcast via LDS
// ---------------------------------------------------------------------------

typedef __attribute__((ext_vector_type(16))) __bf16 v16bf;
typedef __attribute__((ext_vector_type(8)))  __bf16 v8bf;
typedef __attribute__((ext_vector_type(8)))  float  v8f;

#define NWIN    8        // 2 batches * 2x2 windows
#define SEQL    900      // 30*30
#define MROW    (NWIN*SEQL)   // 7200 rows, multiple of 32
#define CMOD    192
#define DIN     384
#define DX2     768      // 2*DIN
#define NPAD    64       // dt_rank + 2*d_state (44) padded to 32-multiple
#define DTRANK  12
#define NSTATE  16
#define NLAYER  12
#define EPSV    1e-5f

// ------------------------------ device helpers -----------------------------

static __device__ __forceinline__ float silu_f(float x) {
    return x / (1.0f + __expf(-x));
}

// ------------------------------ weight converts ----------------------------

__global__ void f32_to_bf16_kernel(const float* __restrict__ src,
                                   __bf16* __restrict__ dst, int n) {
    int i = blockIdx.x * blockDim.x + threadIdx.x;
    if (i < n) dst[i] = (__bf16)src[i];
}

// W_x (12,44,384) -> (12,64,384) zero-padded rows 44..63
__global__ void convert_wx_pad_kernel(const float* __restrict__ src,
                                      __bf16* __restrict__ dst) {
    int i = blockIdx.x * blockDim.x + threadIdx.x;
    const int total = NLAYER * NPAD * DIN;
    if (i >= total) return;
    int layer = i / (NPAD * DIN);
    int rr    = (i / DIN) % NPAD;
    int c     = i % DIN;
    float v = (rr < 44) ? src[((size_t)layer * 44 + rr) * DIN + c] : 0.0f;
    dst[i] = (__bf16)v;
}

// ------------------------------ window partition ---------------------------
// x (2,1,192,60,60) -> hidden (8,900,192); resid = 0
__global__ void partition_kernel(const float* __restrict__ x,
                                 float* __restrict__ hidden,
                                 float* __restrict__ resid) {
    int i = blockIdx.x * blockDim.x + threadIdx.x;
    const int total = NWIN * SEQL * CMOD;
    if (i >= total) return;
    int c  = i % CMOD;
    int l  = (i / CMOD) % SEQL;
    int wb = i / (CMOD * SEQL);
    int b  = wb >> 2;
    int wi = (wb >> 1) & 1;
    int wj = wb & 1;
    int h  = l / 30;
    int w  = l % 30;
    float v = x[((size_t)b * CMOD + c) * 3600 + (size_t)(wi * 30 + h) * 60 + (wj * 30 + w)];
    hidden[i] = v;
    resid[i]  = 0.0f;
}

// ------------------------------ fused add + RMSNorm ------------------------
// one wave32 per row; 192 = 6 channels per lane; writes resid (f32) and h (bf16)
__global__ void addnorm_kernel(float* __restrict__ resid,
                               const float* __restrict__ hidden,
                               const float* __restrict__ nw,
                               __bf16* __restrict__ hbf) {
    int wave = blockIdx.x * (blockDim.x >> 5) + (threadIdx.x >> 5);
    int lane = threadIdx.x & 31;
    if (wave >= MROW) return;
    size_t base = (size_t)wave * CMOD;
    float vals[6];
    float ss = 0.0f;
#pragma unroll
    for (int k = 0; k < 6; ++k) {
        int c = lane + 32 * k;
        float v = resid[base + c] + hidden[base + c];
        resid[base + c] = v;
        vals[k] = v;
        ss += v * v;
    }
#pragma unroll
    for (int m = 16; m >= 1; m >>= 1) ss += __shfl_xor(ss, m, 32);
    float rs = rsqrtf(ss * (1.0f / CMOD) + EPSV);
#pragma unroll
    for (int k = 0; k < 6; ++k) {
        int c = lane + 32 * k;
        hbf[base + c] = (__bf16)(vals[k] * rs * nw[c]);
    }
}

// ------------------------------ WMMA GEMM, 2x2 tiles -----------------------
// C[M,N] = A[M,K] * B[N,K]^T  (A,B bf16 row-major K-contiguous, C f32)
// One wave computes a 32x32 output block = 2x2 WMMA tiles: per K-step,
// 8x global_load_b128 feed 4x v_wmma_f32_16x16x32_bf16 (2 loads/WMMA vs 4
// for the naive tiling), with each A/B fragment reused twice in registers.
// Fragment layouts per CDNA5 ISA 7.12.2 (16-bit A/B, wave32):
//   A 16x32: lanes 0-15 row M=lane, elems 0..7 = K 0..7,  8..15 = K 16..23
//            lanes 16-31 same rows,  elems 0..7 = K 8..15, 8..15 = K 24..31
//   B 32x16: lane (mod 16) = column N, K-half selected by lane>=16.
//   C/D:     lanes 0-15 = M rows 0..7, lanes 16-31 = M rows 8..15.
static __device__ __forceinline__ v16bf pack16(v8bf lo, v8bf hi) {
    v16bf r;
#pragma unroll
    for (int i = 0; i < 8; ++i) { r[i] = lo[i]; r[i + 8] = hi[i]; }
    return r;
}

__global__ void gemm_bf16_wmma_kernel(const __bf16* __restrict__ A,
                                      const __bf16* __restrict__ B,
                                      float* __restrict__ C,
                                      int M, int N, int K) {
    int wave = blockIdx.x * (blockDim.x >> 5) + (threadIdx.x >> 5);
    int lane = threadIdx.x & 31;
    int tilesN = N >> 5;                 // 32-col blocks
    int tilesM = M >> 5;                 // 32-row blocks
    if (wave >= tilesM * tilesN) return;
    int tm = wave / tilesN;
    int tn = wave - tm * tilesN;
    int half = lane >> 4;                // K-half select
    int l16  = lane & 15;

    const __bf16* Ap0 = A + (size_t)(tm * 32 + l16) * K + half * 8;
    const __bf16* Ap1 = Ap0 + (size_t)16 * K;
    const __bf16* Bp0 = B + (size_t)(tn * 32 + l16) * K + half * 8;
    const __bf16* Bp1 = Bp0 + (size_t)16 * K;

    v8f acc00 = {}, acc01 = {}, acc10 = {}, acc11 = {};
    for (int k = 0; k < K; k += 32) {
        v8bf a0lo = *(const v8bf*)(Ap0 + k);
        v8bf a0hi = *(const v8bf*)(Ap0 + k + 16);
        v8bf a1lo = *(const v8bf*)(Ap1 + k);
        v8bf a1hi = *(const v8bf*)(Ap1 + k + 16);
        v8bf b0lo = *(const v8bf*)(Bp0 + k);
        v8bf b0hi = *(const v8bf*)(Bp0 + k + 16);
        v8bf b1lo = *(const v8bf*)(Bp1 + k);
        v8bf b1hi = *(const v8bf*)(Bp1 + k + 16);
        __builtin_prefetch(Ap0 + k + 256, 0, 0);   // global_prefetch_b8
        __builtin_prefetch(Ap1 + k + 256, 0, 0);
        v16bf a0 = pack16(a0lo, a0hi);
        v16bf a1 = pack16(a1lo, a1hi);
        v16bf b0 = pack16(b0lo, b0hi);
        v16bf b1 = pack16(b1lo, b1hi);
        acc00 = __builtin_amdgcn_wmma_f32_16x16x32_bf16(
            false, a0, false, b0, (short)0, acc00, false, false);
        acc01 = __builtin_amdgcn_wmma_f32_16x16x32_bf16(
            false, a0, false, b1, (short)0, acc01, false, false);
        acc10 = __builtin_amdgcn_wmma_f32_16x16x32_bf16(
            false, a1, false, b0, (short)0, acc10, false, false);
        acc11 = __builtin_amdgcn_wmma_f32_16x16x32_bf16(
            false, a1, false, b1, (short)0, acc11, false, false);
    }

    float* Cp = C + (size_t)(tm * 32 + half * 8) * N + tn * 32 + l16;
#pragma unroll
    for (int r = 0; r < 8; ++r) {
        Cp[(size_t)r * N]            = acc00[r];
        Cp[(size_t)r * N + 16]       = acc01[r];
        Cp[(size_t)(r + 16) * N]      = acc10[r];
        Cp[(size_t)(r + 16) * N + 16] = acc11[r];
    }
}

// ------------------------------ causal depthwise conv + SiLU ---------------
// u_pre lives in xz[:, 0:384]; causal within each 900-token window
__global__ void conv_silu_kernel(const float* __restrict__ xz,
                                 const float* __restrict__ cw,   // (384,4)
                                 const float* __restrict__ cb,   // (384,)
                                 float* __restrict__ u,
                                 __bf16* __restrict__ ubf) {
    int i = blockIdx.x * blockDim.x + threadIdx.x;
    if (i >= MROW * DIN) return;
    int row = i / DIN;
    int d   = i - row * DIN;
    int l   = row % SEQL;
    float acc = cb[d];
#pragma unroll
    for (int t = 0; t < 4; ++t) {
        int ll = l - 3 + t;
        if (ll >= 0)
            acc += xz[(size_t)(row - 3 + t) * DX2 + d] * cw[d * 4 + t];
    }
    float s = silu_f(acc);
    u[i]   = s;
    ubf[i] = (__bf16)s;
}

// ------------------------------ dt = softplus(dtr @ W_dt^T + bias) ---------
__global__ void dt_kernel(const float* __restrict__ dbl,   // (M,64): [0:12]=dtr
                          const float* __restrict__ Wdt,   // (384,12)
                          const float* __restrict__ dtb,   // (384,)
                          float* __restrict__ dt) {
    int i = blockIdx.x * blockDim.x + threadIdx.x;
    if (i >= MROW * DIN) return;
    int row = i / DIN;
    int d   = i - row * DIN;
    float acc = dtb[d];
    const float* dr = dbl + (size_t)row * NPAD;
    const float* wd = Wdt + d * DTRANK;
#pragma unroll
    for (int r = 0; r < DTRANK; ++r) acc += dr[r] * wd[r];
    dt[i] = (acc > 20.0f) ? acc : log1pf(__expf(acc));
}

// ------------------------------ selective scan -----------------------------
// 24 blocks = 8 windows x 3 chunks of 128 channels. 16 states / thread in
// VGPRs; B/C (32 floats, shared across channels of a window) staged in LDS.
// Fuses y*silu(z) and the bf16 store feeding the output GEMM.
__global__ void scan_kernel(const float* __restrict__ u,
                            const float* __restrict__ dt,
                            const float* __restrict__ dbl,  // (M,64): B@12, C@28
                            const float* __restrict__ xz,   // z at [:,384:768]
                            const float* __restrict__ Alog, // (384,16)
                            const float* __restrict__ Dsk,  // (384,)
                            __bf16* __restrict__ yzbf) {
    __shared__ float bc[2 * NSTATE];
    int wb    = blockIdx.x / 3;
    int chunk = blockIdx.x % 3;
    int d     = chunk * 128 + threadIdx.x;

    float Arow[NSTATE], h[NSTATE];
    float Dd = Dsk[d];
#pragma unroll
    for (int n = 0; n < NSTATE; ++n) {
        Arow[n] = -__expf(Alog[d * NSTATE + n]);
        h[n]    = 0.0f;
    }
    for (int l = 0; l < SEQL; ++l) {
        int row = wb * SEQL + l;
        if (threadIdx.x < 32)
            bc[threadIdx.x] = dbl[(size_t)row * NPAD + DTRANK + threadIdx.x];
        __syncthreads();
        float dtv = dt[(size_t)row * DIN + d];
        float uv  = u[(size_t)row * DIN + d];
        float y   = uv * Dd;
#pragma unroll
        for (int n = 0; n < NSTATE; ++n) {
            float dA = __expf(dtv * Arow[n]);           // v_exp_f32
            h[n] = dA * h[n] + (dtv * uv) * bc[n];
            y += h[n] * bc[NSTATE + n];
        }
        float z = xz[(size_t)row * DX2 + DIN + d];
        yzbf[(size_t)row * DIN + d] = (__bf16)(y * silu_f(z));
        __syncthreads();
    }
}

// ------------------------------ final add + RMSNorm + unpartition ----------
__global__ void final_kernel(const float* __restrict__ resid,
                             const float* __restrict__ hidden,
                             const float* __restrict__ nfw,
                             float* __restrict__ out) {
    int wave = blockIdx.x * (blockDim.x >> 5) + (threadIdx.x >> 5);
    int lane = threadIdx.x & 31;
    if (wave >= MROW) return;
    size_t base = (size_t)wave * CMOD;
    int wb = wave / SEQL;
    int l  = wave % SEQL;
    int b  = wb >> 2;
    int wi = (wb >> 1) & 1;
    int wj = wb & 1;
    int hh = l / 30;
    int ww = l % 30;
    size_t spatial = (size_t)(wi * 30 + hh) * 60 + (wj * 30 + ww);

    float vals[6];
    float ss = 0.0f;
#pragma unroll
    for (int k = 0; k < 6; ++k) {
        int c = lane + 32 * k;
        float v = resid[base + c] + hidden[base + c];
        vals[k] = v;
        ss += v * v;
    }
#pragma unroll
    for (int m = 16; m >= 1; m >>= 1) ss += __shfl_xor(ss, m, 32);
    float rs = rsqrtf(ss * (1.0f / CMOD) + EPSV);
#pragma unroll
    for (int k = 0; k < 6; ++k) {
        int c = lane + 32 * k;
        out[((size_t)b * CMOD + c) * 3600 + spatial] = vals[k] * rs * nfw[c];
    }
}

// ---------------------------------------------------------------------------

extern "C" void kernel_launch(void* const* d_in, const int* in_sizes, int n_in,
                              void* d_out, int out_size, void* d_ws, size_t ws_size,
                              hipStream_t stream) {
    (void)in_sizes; (void)n_in; (void)out_size; (void)ws_size;

    const float* x       = (const float*)d_in[0];   // (2,1,192,60,60)
    const float* W_in    = (const float*)d_in[1];   // (12,768,192)
    const float* conv_w  = (const float*)d_in[2];   // (12,384,4)
    const float* conv_b  = (const float*)d_in[3];   // (12,384)
    const float* W_x     = (const float*)d_in[4];   // (12,44,384)
    const float* W_dt    = (const float*)d_in[5];   // (12,384,12)
    const float* dt_bias = (const float*)d_in[6];   // (12,384)
    const float* A_log   = (const float*)d_in[7];   // (12,384,16)
    const float* D_skip  = (const float*)d_in[8];   // (12,384)
    const float* W_out   = (const float*)d_in[9];   // (12,192,384)
    const float* norm_w  = (const float*)d_in[10];  // (12,192)
    const float* norm_fw = (const float*)d_in[11];  // (192,)
    float* out = (float*)d_out;

    // ---- workspace carve (256B aligned) ----
    char* base = (char*)d_ws;
    size_t off = 0;
    auto carve = [&](size_t nbytes) -> void* {
        void* p = base + off;
        off += (nbytes + 255) & ~(size_t)255;
        return p;
    };
    float*  resid  = (float*)carve((size_t)MROW * CMOD * 4);
    float*  hidden = (float*)carve((size_t)MROW * CMOD * 4);
    __bf16* hbf    = (__bf16*)carve((size_t)MROW * CMOD * 2);
    float*  xz     = (float*)carve((size_t)MROW * DX2 * 4);
    float*  u      = (float*)carve((size_t)MROW * DIN * 4);
    __bf16* ubf    = (__bf16*)carve((size_t)MROW * DIN * 2);
    float*  dbl    = (float*)carve((size_t)MROW * NPAD * 4);
    float*  dt     = (float*)carve((size_t)MROW * DIN * 4);
    __bf16* yzbf   = (__bf16*)carve((size_t)MROW * DIN * 2);
    __bf16* Wi     = (__bf16*)carve((size_t)NLAYER * DX2 * CMOD * 2);
    __bf16* Wx     = (__bf16*)carve((size_t)NLAYER * NPAD * DIN * 2);
    __bf16* Wo     = (__bf16*)carve((size_t)NLAYER * CMOD * DIN * 2);

    const int TB = 256;
    auto cdiv = [](int a, int b) { return (a + b - 1) / b; };

    // ---- weight conversion to bf16 (once per launch) ----
    {
        int n1 = NLAYER * DX2 * CMOD;
        f32_to_bf16_kernel<<<cdiv(n1, TB), TB, 0, stream>>>(W_in, Wi, n1);
        int n2 = NLAYER * NPAD * DIN;
        convert_wx_pad_kernel<<<cdiv(n2, TB), TB, 0, stream>>>(W_x, Wx);
        int n3 = NLAYER * CMOD * DIN;
        f32_to_bf16_kernel<<<cdiv(n3, TB), TB, 0, stream>>>(W_out, Wo, n3);
    }

    // ---- window partition; hidden = xw, resid = 0 ----
    partition_kernel<<<cdiv(NWIN * SEQL * CMOD, TB), TB, 0, stream>>>(x, hidden, resid);

    const int rowsBlocks = cdiv(MROW, TB / 32);            // wave-per-row kernels
    const int gXZ  = cdiv((MROW / 32) * (DX2 / 32), TB / 32);
    const int gDBL = cdiv((MROW / 32) * (NPAD / 32), TB / 32);
    const int gOUT = cdiv((MROW / 32) * (CMOD / 32), TB / 32);

    for (int layer = 0; layer < NLAYER; ++layer) {
        // resid += hidden ; h = rmsnorm(resid) * norm_w -> bf16
        addnorm_kernel<<<rowsBlocks, TB, 0, stream>>>(
            resid, hidden, norm_w + (size_t)layer * CMOD, hbf);

        // xz = h @ W_in^T  (7200x192 x 192x768)
        gemm_bf16_wmma_kernel<<<gXZ, TB, 0, stream>>>(
            hbf, Wi + (size_t)layer * DX2 * CMOD, xz, MROW, DX2, CMOD);

        // u = silu(depthwise_causal_conv(xz[:, :384]))
        conv_silu_kernel<<<cdiv(MROW * DIN, TB), TB, 0, stream>>>(
            xz, conv_w + (size_t)layer * DIN * 4, conv_b + (size_t)layer * DIN,
            u, ubf);

        // dbl = u @ W_x^T  (7200x384 x 384x64[padded])
        gemm_bf16_wmma_kernel<<<gDBL, TB, 0, stream>>>(
            ubf, Wx + (size_t)layer * NPAD * DIN, dbl, MROW, NPAD, DIN);

        // dt = softplus(dtr @ W_dt^T + dt_bias)
        dt_kernel<<<cdiv(MROW * DIN, TB), TB, 0, stream>>>(
            dbl, W_dt + (size_t)layer * DIN * DTRANK,
            dt_bias + (size_t)layer * DIN, dt);

        // selective scan + y*silu(z) -> bf16
        scan_kernel<<<NWIN * 3, 128, 0, stream>>>(
            u, dt, dbl, xz, A_log + (size_t)layer * DIN * NSTATE,
            D_skip + (size_t)layer * DIN, yzbf);

        // hidden = yz @ W_out^T  (7200x384 x 384x192)
        gemm_bf16_wmma_kernel<<<gOUT, TB, 0, stream>>>(
            yzbf, Wo + (size_t)layer * CMOD * DIN, hidden, MROW, CMOD, DIN);
    }

    // final add + RMSNorm + window unpartition
    final_kernel<<<rowsBlocks, TB, 0, stream>>>(resid, hidden, norm_fw, out);
}
